// SparseBlock_17085379903854
// MI455X (gfx1250) — matrix operand order repaired and checked by reading, not compiled, not run on hardware
//
#include <hip/hip_runtime.h>

typedef __bf16 bf16_t;
typedef __attribute__((ext_vector_type(16))) __bf16 v16bf;
typedef __attribute__((ext_vector_type(8)))  float  v8f;

#define B_    4
#define CIN_  64
#define COUT_ 128
#define H_    512
#define W_    512
#define HW_   262144LL
#define NBLK_ 512
#define NTILE_ 1024          // B * GH * GW = 4*16*16
#define EPS_  1e-5f

// workspace layout (bytes)
#define WS_W1B  0                    // 128*64 bf16   = 16 KB
#define WS_WDB  16384                // 128*64 bf16   = 16 KB
#define WS_W3B  32768                // 128*128 bf16  = 32 KB
#define WS_W2B  65536                // 9*128*128 bf16 = 288 KB, [tap][co][ci]
#define WS_BN   (65536 + 294912)     // 6*128 f32 (s1,t1,s2,t2,s3,t3)
#define WS_LUT  (WS_BN + 3072)       // 1024 int (tile -> active?)

static __device__ __forceinline__ v8f wmma_bf16(v16bf a, v16bf b, v8f c) {
  // (neg_a, A, neg_b, B, c_mod, C, reuse_a, reuse_b)
  return __builtin_amdgcn_wmma_f32_16x16x32_bf16(false, a, false, b, (short)0, c,
                                                 false, false);
}

// A-matrix 16x32 bf16 fragment, ISA 7.12.2: lane m=lane&15, half h=lane>>4.
// elems 0..7  -> k = 8h + e         (contiguous 16B run)
// elems 8..15 -> k = 16 + 8h + (e-8) (contiguous 16B run)
static __device__ __forceinline__ v16bf load_a_frag(const bf16_t* wrow, int h) {
  union { v16bf v; uint4 u[2]; } A;
  A.u[0] = *(const uint4*)(wrow + 8 * h);
  A.u[1] = *(const uint4*)(wrow + 16 + 8 * h);
  return A.v;
}

// B-matrix 32x16 bf16 fragment: lane n=lane&15, half h=lane>>4; k = 8h + e (e<8), 16 + 8h + (e-8) (e>=8).
static __device__ __forceinline__ v16bf load_b_frag(const bf16_t* row, int h) {
  union { v16bf v; uint4 u[2]; } Bf;
  Bf.u[0] = *(const uint4*)(row + 8 * h);
  Bf.u[1] = *(const uint4*)(row + 16 + 8 * h);
  return Bf.v;
}

static __device__ __forceinline__ v16bf zero_frag() {
  union { v16bf v; uint4 u[2]; } Z;
  Z.u[0] = make_uint4(0u, 0u, 0u, 0u);
  Z.u[1] = Z.u[0];
  return Z.v;
}

static __device__ __forceinline__ v8f zero_acc() {
  v8f z = {0.f, 0.f, 0.f, 0.f, 0.f, 0.f, 0.f, 0.f};
  return z;
}

// ---------------------------------------------------------------------------
// Kernel 0: weights -> bf16, fold conv-bias + BN into scale/shift, build the
// active-tile LUT (race-free: each tile scans all 512 indices).
// ---------------------------------------------------------------------------
__global__ __launch_bounds__(256) void prep_kernel(
    const int* __restrict__ idx,
    const float* __restrict__ w1, const float* __restrict__ b1,
    const float* __restrict__ w2, const float* __restrict__ b2,
    const float* __restrict__ w3, const float* __restrict__ b3,
    const float* __restrict__ wd,
    const float* __restrict__ g1, const float* __restrict__ be1,
    const float* __restrict__ m1, const float* __restrict__ v1,
    const float* __restrict__ g2, const float* __restrict__ be2,
    const float* __restrict__ m2, const float* __restrict__ v2,
    const float* __restrict__ g3, const float* __restrict__ be3,
    const float* __restrict__ m3, const float* __restrict__ v3,
    bf16_t* __restrict__ w1b, bf16_t* __restrict__ wdb,
    bf16_t* __restrict__ w3b, bf16_t* __restrict__ w2b,
    float* __restrict__ bn, int* __restrict__ lut) {
  const int stride = gridDim.x * blockDim.x;
  const int tid0 = blockIdx.x * blockDim.x + threadIdx.x;
  for (int t = tid0; t < COUT_ * CIN_; t += stride) {
    w1b[t] = (bf16_t)w1[t];
    wdb[t] = (bf16_t)wd[t];
  }
  for (int t = tid0; t < COUT_ * COUT_; t += stride) w3b[t] = (bf16_t)w3[t];
  for (int t = tid0; t < 9 * COUT_ * COUT_; t += stride) {
    int tap = t / (COUT_ * COUT_);
    int rem = t - tap * (COUT_ * COUT_);
    int co = rem >> 7, ci = rem & 127;
    int ky = tap / 3, kx = tap % 3;
    w2b[t] = (bf16_t)w2[((co * COUT_ + ci) * 3 + ky) * 3 + kx];
  }
  for (int t = tid0; t < COUT_; t += stride) {
    float s = g1[t] * rsqrtf(v1[t] + EPS_);
    bn[t] = s;
    bn[COUT_ + t] = (b1[t] - m1[t]) * s + be1[t];
    s = g2[t] * rsqrtf(v2[t] + EPS_);
    bn[2 * COUT_ + t] = s;
    bn[3 * COUT_ + t] = (b2[t] - m2[t]) * s + be2[t];
    s = g3[t] * rsqrtf(v3[t] + EPS_);
    bn[4 * COUT_ + t] = s;
    bn[5 * COUT_ + t] = (b3[t] - m3[t]) * s + be3[t];
  }
  for (int t = tid0; t < NTILE_; t += stride) {
    int act = 0;
    for (int j = 0; j < NBLK_; ++j) {
      int flat = idx[j * 3 + 0] * 256 + idx[j * 3 + 1] * 16 + idx[j * 3 + 2];
      act |= (flat == t) ? 1 : 0;
    }
    lut[t] = act;
  }
}

// ---------------------------------------------------------------------------
// Fused kernel: one WG (256 thr / 8 waves) per 32x32 grid tile.
//   - every tile: ybase = wd . x + bd  (bf16 WMMA GEMM from LDS-staged x)
//   - active tiles: + relu(bn3(conv3(relu(bn2(conv2(relu(bn1(conv1(x)))))))))
// Single write of out (no RMW, x staged once incl. conv2 halo).
// Strip-mined: 8 strips of 4 output rows; d1 staged rows r0-1..r0+4; conv2
// halo rows/cols are literal ZERO (reference pads conv2 input with zeros).
// ---------------------------------------------------------------------------
__global__ __launch_bounds__(256) void fused_tile_kernel(
    const float* __restrict__ x, const int* __restrict__ lut,
    const bf16_t* __restrict__ wdb, const bf16_t* __restrict__ w1b,
    const bf16_t* __restrict__ w2b, const bf16_t* __restrict__ w3b,
    const float* __restrict__ bn, const float* __restrict__ bd,
    float* __restrict__ out) {
  __shared__ bf16_t d1s[6 * 32 * COUT_];  // [staged_row*32+col][co], 48 KB
  __shared__ bf16_t xs[96 * CIN_];        // [px][ci], 12 KB (3-row subpasses)
  bf16_t* d2s = d1s;                      // d2 (32 KB) aliases d1 region

  const int tid = threadIdx.x;
  const int lane = tid & 31, wv = tid >> 5;
  const int l15 = lane & 15, h = lane >> 4;
  const int tile = blockIdx.x;
  const int b = tile >> 8, gh = (tile >> 4) & 15, gw = tile & 15;
  const bool active = lut[tile] != 0;  // uniform per WG -> scalar branch

  const float* s1 = bn;             const float* t1 = bn + COUT_;
  const float* s2 = bn + 2 * COUT_; const float* t2 = bn + 3 * COUT_;
  const float* s3 = bn + 4 * COUT_; const float* t3 = bn + 5 * COUT_;

  for (int r0 = 0; r0 < 32; r0 += 4) {
    if (tid < CIN_ && r0 + 5 < 32)
      __builtin_prefetch(&x[((long long)b * CIN_ + tid) * HW_ +
                            (long long)(gh * 32 + r0 + 5) * W_ + gw * 32],
                         0, 0);

    v8f accY[8];  // ybase accumulators, live across the whole strip
#pragma unroll
    for (int nt = 0; nt < 8; ++nt) accY[nt] = zero_acc();

    // ---- two 3-row staging subpasses: sp0 -> sr0..2, sp1 -> sr3..5 ----
    for (int sp = 0; sp < 2; ++sp) {
      __syncthreads();  // fences prev strip's d2s reads vs d1s writes too
      for (int t = tid; t < 96 * CIN_; t += 256) {
        int ci = t / 96, p = t - ci * 96;
        int sr = sp * 3 + (p >> 5);
        int br = r0 - 1 + sr;
        int col = p & 31;
        bool need = active || (sr >= 1 && sr <= 4);  // inactive: ybase rows only
        float v = 0.f;
        if (need && br >= 0 && br < 32)
          v = x[((long long)b * CIN_ + ci) * HW_ +
                (long long)(gh * 32 + br) * W_ + gw * 32 + col];
        xs[p * CIN_ + ci] = (bf16_t)v;
      }
      __syncthreads();

      // ybase partial GEMM: sp0 covers strip rows 0..1 (ntiles 0..3),
      // sp1 covers strip rows 2..3 (ntiles 4..7).
      for (int q = 0; q < 4; ++q) {
        const int nt = sp * 4 + q;
        const int p = nt * 16 + l15;          // strip pixel 0..127
        const int orow = p >> 5, ocol = p & 31;
        const int xsp = (orow + 1 - 3 * sp) * 32 + ocol;  // row in xs
        for (int kk = 0; kk < CIN_; kk += 32) {
          v16bf a = load_a_frag(wdb + (wv * 16 + l15) * CIN_ + kk, h);
          v16bf bb = load_b_frag(xs + xsp * CIN_ + kk, h);
          accY[nt] = wmma_bf16(a, bb, accY[nt]);
        }
      }

      // conv1 -> d1s (active tiles only; uniform branch, EXEC stays full)
      if (active) {
        for (int nt = 0; nt < 6; ++nt) {
          v8f acc = zero_acc();
          const int p = nt * 16 + l15;
          for (int kk = 0; kk < CIN_; kk += 32) {
            v16bf a = load_a_frag(w1b + (wv * 16 + l15) * CIN_ + kk, h);
            v16bf bb = load_b_frag(xs + p * CIN_ + kk, h);
            acc = wmma_bf16(a, bb, acc);
          }
          int spx = sp * 96 + p;
          int sr = spx >> 5, col = spx & 31;
          int br = r0 - 1 + sr;
          bool valid = (br >= 0) && (br < 32);  // halo rows = literal zero
#pragma unroll
          for (int r = 0; r < 8; ++r) {
            int co = wv * 16 + r + 8 * h;
            float v = valid ? fmaxf(acc[r] * s1[co] + t1[co], 0.f) : 0.f;
            d1s[(sr * 32 + col) * COUT_ + co] = (bf16_t)v;
          }
        }
      }
    }
    __syncthreads();

    if (active) {
      // ---- conv2: implicit GEMM, 9 taps x K=128 over 4x32 output px ----
      v8f acc2[8];
#pragma unroll
      for (int nt = 0; nt < 8; ++nt) acc2[nt] = zero_acc();
      for (int tap = 0; tap < 9; ++tap) {
        const int dy = tap / 3;       // staged source row = orow + dy (0..5)
        const int dx = tap % 3 - 1;
        const bf16_t* wt =
            w2b + (long long)tap * COUT_ * COUT_ + (wv * 16 + l15) * COUT_;
        for (int kk = 0; kk < COUT_; kk += 32) {
          v16bf a = load_a_frag(wt + kk, h);  // reused across 8 ntiles
#pragma unroll
          for (int nt = 0; nt < 8; ++nt) {
            int p = nt * 16 + l15;
            int orow = p >> 5, ocol = p & 31;
            int scol = ocol + dx;
            v16bf bb;
            if (scol >= 0 && scol < 32)
              bb = load_b_frag(d1s + ((orow + dy) * 32 + scol) * COUT_ + kk, h);
            else
              bb = zero_frag();       // SAME padding at block edge
            acc2[nt] = wmma_bf16(a, bb, acc2[nt]);
          }
        }
      }
      __syncthreads();  // all d1s reads done before aliased d2s writes

      // bn2 + relu -> d2s
#pragma unroll
      for (int nt = 0; nt < 8; ++nt) {
        int p = nt * 16 + l15;
#pragma unroll
        for (int r = 0; r < 8; ++r) {
          int co = wv * 16 + r + 8 * h;
          float v = fmaxf(acc2[nt][r] * s2[co] + t2[co], 0.f);
          d2s[p * COUT_ + co] = (bf16_t)v;
        }
      }
      __syncthreads();

      // conv3 (1x1) + bn3 + relu, folded into accY
      for (int nt = 0; nt < 8; ++nt) {
        v8f acc = zero_acc();
        const int p = nt * 16 + l15;
        for (int kk = 0; kk < COUT_; kk += 32) {
          v16bf a = load_a_frag(w3b + (wv * 16 + l15) * COUT_ + kk, h);
          v16bf bb = load_b_frag(d2s + p * COUT_ + kk, h);
          acc = wmma_bf16(a, bb, acc);
        }
#pragma unroll
        for (int r = 0; r < 8; ++r) {
          int co = wv * 16 + r + 8 * h;
          accY[nt][r] += fmaxf(acc[r] * s3[co] + t3[co], 0.f);
        }
      }
    }

    // ---- single store: out = ybase + bd (+ scatter contribution) ----
    for (int nt = 0; nt < 8; ++nt) {
      const int p = nt * 16 + l15;
      const int row = r0 + (p >> 5), col = p & 31;
#pragma unroll
      for (int r = 0; r < 8; ++r) {
        int co = wv * 16 + r + 8 * h;  // C/D layout: m = r + 8h
        out[((long long)b * COUT_ + co) * HW_ +
            (long long)(gh * 32 + row) * W_ + gw * 32 + col] =
            accY[nt][r] + bd[co];
      }
    }
    // next strip's first __syncthreads fences d2s reads vs d1s writes
  }
}

// ---------------------------------------------------------------------------
extern "C" void kernel_launch(void* const* d_in, const int* in_sizes, int n_in,
                              void* d_out, int out_size, void* d_ws,
                              size_t ws_size, hipStream_t stream) {
  (void)in_sizes; (void)n_in; (void)out_size; (void)ws_size;
  const float* x  = (const float*)d_in[0];
  const int* idx  = (const int*)d_in[1];
  const float* w1 = (const float*)d_in[2];
  const float* b1 = (const float*)d_in[3];
  const float* w2 = (const float*)d_in[4];
  const float* b2 = (const float*)d_in[5];
  const float* w3 = (const float*)d_in[6];
  const float* b3 = (const float*)d_in[7];
  const float* wd = (const float*)d_in[8];
  const float* bd = (const float*)d_in[9];
  const float* g1 = (const float*)d_in[10];
  const float* be1 = (const float*)d_in[11];
  const float* m1 = (const float*)d_in[12];
  const float* v1 = (const float*)d_in[13];
  const float* g2 = (const float*)d_in[14];
  const float* be2 = (const float*)d_in[15];
  const float* m2 = (const float*)d_in[16];
  const float* v2 = (const float*)d_in[17];
  const float* g3 = (const float*)d_in[18];
  const float* be3 = (const float*)d_in[19];
  const float* m3 = (const float*)d_in[20];
  const float* v3 = (const float*)d_in[21];
  float* out = (float*)d_out;

  char* ws = (char*)d_ws;
  bf16_t* w1b = (bf16_t*)(ws + WS_W1B);
  bf16_t* wdb = (bf16_t*)(ws + WS_WDB);
  bf16_t* w3b = (bf16_t*)(ws + WS_W3B);
  bf16_t* w2b = (bf16_t*)(ws + WS_W2B);
  float* bnp = (float*)(ws + WS_BN);
  int* lutp = (int*)(ws + WS_LUT);

  prep_kernel<<<512, 256, 0, stream>>>(idx, w1, b1, w2, b2, w3, b3, wd, g1,
                                       be1, m1, v1, g2, be2, m2, v2, g3, be3,
                                       m3, v3, w1b, wdb, w3b, w2b, bnp, lutp);
  fused_tile_kernel<<<NTILE_, 256, 0, stream>>>(x, lutp, wdb, w1b, w2b, w3b,
                                                bnp, bd, out);
}
